// Attention_55362128445856
// MI455X (gfx1250) — compile-verified, matrix operand order checked by reading, not run on hardware
//
#include <hip/hip_runtime.h>

typedef __attribute__((ext_vector_type(16))) __bf16 v16bf;
typedef __attribute__((ext_vector_type(8)))  float  v8f;
typedef unsigned short u16;
typedef unsigned int   u32;

#define BATCH 8
#define SEQ   1024
#define CH    768
#define HEADS 12
#define DH    64
#define NQKV  2304
#define MROWS (BATCH*SEQ)   // 8192
#define LORA_SCALE 8.0f
#define ATT_SCALE  0.125f   // 1/sqrt(64)

union AFrag { v16bf v; uint4 q[2]; };
union FAcc  { v8f   v; float f[8]; };

__device__ __forceinline__ u16 f2bf(float x) {
  union { float f; u32 u; } c; c.f = x;
  u32 r = c.u + 0x7FFFu + ((c.u >> 16) & 1u);
  return (u16)(r >> 16);
}

__device__ __forceinline__ v8f wmma_bf16(const v16bf& a, const v16bf& b, const v8f& c) {
  return __builtin_amdgcn_wmma_f32_16x16x32_bf16(false, a, false, b, (short)0, c, false, false);
}

// ---------------------------------------------------------------- cast f32->bf16 (row-major copy)
__global__ void cast_bf16_kernel(const float* __restrict__ src, u16* __restrict__ dst, size_t n) {
  size_t i = (size_t)blockIdx.x * blockDim.x + threadIdx.x;
  size_t stride = (size_t)gridDim.x * blockDim.x;
  for (; i < n; i += stride) dst[i] = f2bf(src[i]);
}

// ---------------------------------------------------------------- cast+transpose: src[K][N] f32 -> dst[N][K] bf16
__global__ void cast_transpose_kernel(const float* __restrict__ src, u16* __restrict__ dst,
                                      int K, int N) {
  size_t total = (size_t)K * N;
  size_t i = (size_t)blockIdx.x * blockDim.x + threadIdx.x;
  size_t stride = (size_t)gridDim.x * blockDim.x;
  for (; i < total; i += stride) {
    int n = (int)(i / K);
    int k = (int)(i % K);
    dst[i] = f2bf(src[(size_t)k * N + n]);
  }
}

// ---------------------------------------------------------------- t[row][0..3] = src[row,:] @ A[:,0..3]
__global__ void rank4_kernel(const float* __restrict__ src, const float* __restrict__ A,
                             float* __restrict__ t, int C) {
  __shared__ float red[256][4];
  const int row = blockIdx.x;
  const int tid = threadIdx.x;
  float a0 = 0.f, a1 = 0.f, a2 = 0.f, a3 = 0.f;
  const float* srow = src + (size_t)row * C;
  for (int c = tid; c < C; c += 256) {
    float x = srow[c];
    const float* Ar = A + (size_t)c * 4;
    a0 += x * Ar[0]; a1 += x * Ar[1]; a2 += x * Ar[2]; a3 += x * Ar[3];
  }
  red[tid][0] = a0; red[tid][1] = a1; red[tid][2] = a2; red[tid][3] = a3;
  __syncthreads();
  for (int off = 128; off > 0; off >>= 1) {
    if (tid < off)
      for (int q = 0; q < 4; ++q) red[tid][q] += red[tid + off][q];
    __syncthreads();
  }
  if (tid < 4) t[(size_t)row * 4 + tid] = red[0][tid];
}

// ---------------------------------------------------------------- tiled bf16 WMMA GEMM
// C[M][N] = A[M][K](bf16, row-major) @ B (PRE-TRANSPOSED: Bt[N][K] bf16)
// block tile 128x128, BK=64; wave tile 32x64 -> 16 wmma per stage
// register-staged pipeline: next chunk's global loads overlap current compute
// mode 0: += LORA_SCALE * t[M][4] @ Blora[4][N], store bf16 -> outb
// mode 1: += bias[N], store f32 -> outf
__global__ __launch_bounds__(256)
void gemm_bf16_kernel(const u16* __restrict__ A, const u16* __restrict__ Bt,
                      int M, int N, int K, int mode,
                      const float* __restrict__ t, const float* __restrict__ Blora,
                      const float* __restrict__ bias,
                      u16* __restrict__ outb, float* __restrict__ outf) {
  __shared__ __align__(16) u16 As[128 * 72];   // [m][k64], pad->72
  __shared__ __align__(16) u16 Bs[128 * 72];   // [n][k64], pad->72

  const int tid  = threadIdx.x;
  const int lane = tid & 31;
  const int wave = tid >> 5;
  const int lm   = lane & 15;
  const int kh   = lane >> 4;

  const int block_n0 = blockIdx.x * 128;
  const int block_m0 = blockIdx.y * 128;
  const int wm = (wave >> 1) * 32;   // 0,32,64,96
  const int wn = (wave & 1) * 64;    // 0,64

  FAcc acc[2][4];
#pragma unroll
  for (int a = 0; a < 2; ++a)
#pragma unroll
    for (int b = 0; b < 4; ++b)
#pragma unroll
      for (int j = 0; j < 8; ++j) acc[a][b].f[j] = 0.f;

  const int cp = tid & 15;     // uint2 column within 64-elem row (16 uint2/row)
  const int rr = tid >> 4;     // 0..15
  const u16* Abase = A  + (size_t)(block_m0 + rr) * K + cp * 4;
  const u16* Bbase = Bt + (size_t)(block_n0 + rr) * K + cp * 4;

  // preload first chunk into registers (issues as one load clause)
  uint2 ra[8], rb[8];
#pragma unroll
  for (int p = 0; p < 8; ++p) {
    ra[p] = *(const uint2*)(Abase + (size_t)p * 16 * K);
    rb[p] = *(const uint2*)(Bbase + (size_t)p * 16 * K);
  }

  for (int k0 = 0; k0 < K; k0 += 64) {
    __syncthreads();
    // commit current chunk registers -> LDS
#pragma unroll
    for (int p = 0; p < 8; ++p) {
      *(uint2*)(&As[(rr + p * 16) * 72 + cp * 4]) = ra[p];
      *(uint2*)(&Bs[(rr + p * 16) * 72 + cp * 4]) = rb[p];
    }
    // kick off next chunk's global loads; latency hides under compute below
    const int kn = (k0 + 64 < K) ? (k0 + 64) : k0;   // clamped tail (deterministic)
#pragma unroll
    for (int p = 0; p < 8; ++p) {
      ra[p] = *(const uint2*)(Abase + (size_t)p * 16 * K + kn);
      rb[p] = *(const uint2*)(Bbase + (size_t)p * 16 * K + kn);
    }
    __syncthreads();

#pragma unroll
    for (int ks = 0; ks < 2; ++ks) {
      AFrag af[2], bf[4];
#pragma unroll
      for (int ti = 0; ti < 2; ++ti) {
        const u16* p = &As[(wm + ti * 16 + lm) * 72 + ks * 32 + kh * 8];
        af[ti].q[0] = *(const uint4*)(p);
        af[ti].q[1] = *(const uint4*)(p + 16);
      }
#pragma unroll
      for (int tj = 0; tj < 4; ++tj) {
        const u16* p = &Bs[(wn + tj * 16 + lm) * 72 + ks * 32 + kh * 8];
        bf[tj].q[0] = *(const uint4*)(p);
        bf[tj].q[1] = *(const uint4*)(p + 16);
      }
#pragma unroll
      for (int ti = 0; ti < 2; ++ti)
#pragma unroll
        for (int tj = 0; tj < 4; ++tj)
          acc[ti][tj].v = wmma_bf16(af[ti].v, bf[tj].v, acc[ti][tj].v);
    }
  }

  // epilogue
#pragma unroll
  for (int tj = 0; tj < 4; ++tj) {
    const int col = block_n0 + wn + tj * 16 + lm;
    float bl0 = 0.f, bl1 = 0.f, bl2 = 0.f, bl3 = 0.f, bv = 0.f;
    if (mode == 0) {
      bl0 = Blora[0 * N + col]; bl1 = Blora[1 * N + col];
      bl2 = Blora[2 * N + col]; bl3 = Blora[3 * N + col];
    } else {
      bv = bias[col];
    }
#pragma unroll
    for (int ti = 0; ti < 2; ++ti) {
#pragma unroll
      for (int j = 0; j < 8; ++j) {
        const int row = block_m0 + wm + ti * 16 + j + 8 * kh;
        float v = acc[ti][tj].f[j];
        if (mode == 0) {
          const float4 tv = *(const float4*)(t + (size_t)row * 4);
          v += LORA_SCALE * (tv.x * bl0 + tv.y * bl1 + tv.z * bl2 + tv.w * bl3);
          outb[(size_t)row * N + col] = f2bf(v);
        } else {
          outf[(size_t)row * N + col] = v + bv;
        }
      }
    }
  }
}

// ---------------------------------------------------------------- flash attention (bf16 WMMA, f32 softmax)
// qkv: bf16 [B*N][2304] as [q(768) | k(768) | v(768)], head h at offset h*64
// ob : bf16 [B*N][768]
// 64-key chunks: 8 wmma QK^T + 8 wmma PV per stage; register-staged K/V pipeline
__global__ __launch_bounds__(256)
void attn_kernel(const u16* __restrict__ qkv, u16* __restrict__ ob) {
  __shared__ __align__(16) u16 Ks[64 * 72];        // [key][dk64], pad->72
  __shared__ __align__(16) u16 Vt[64 * 72];        // [dv][key64], pad->72
  __shared__ __align__(16) u16 Ps[8 * 16 * 64];    // per-wave P: [m][key64]

  const int bid = blockIdx.x;
  const int qt  = bid & 7;              // 8 query tiles of 128
  const int h   = (bid >> 3) % HEADS;
  const int b   = bid / (8 * HEADS);

  const int tid  = threadIdx.x;
  const int lane = tid & 31;
  const int wave = tid >> 5;
  const int lm   = lane & 15;
  const int kh   = lane >> 4;

  const int q0 = qt * 128 + wave * 16;

  // Q A-fragments (d=64 -> two k-steps of 32), loaded straight from global
  AFrag aq[2];
  {
    const size_t base = ((size_t)(b * SEQ + q0 + lm)) * NQKV + h * DH;
#pragma unroll
    for (int i = 0; i < 2; ++i) {
      const u16* p = qkv + base + i * 32 + kh * 8;
      aq[i].q[0] = *(const uint4*)(p);
      aq[i].q[1] = *(const uint4*)(p + 16);
    }
  }

  float rm[8], rl[8];
  FAcc oacc[4];
#pragma unroll
  for (int j = 0; j < 8; ++j) { rm[j] = -1e30f; rl[j] = 0.f; }
#pragma unroll
  for (int tI = 0; tI < 4; ++tI)
#pragma unroll
    for (int j = 0; j < 8; ++j) oacc[tI].f[j] = 0.f;

  u16* Pw = &Ps[wave * 16 * 64];

  const int cp = tid & 15;   // uint2 col in 64-elem dk row (K staging)
  const int rr = tid >> 4;   // 0..15
  const int np = tid & 31;   // dv pair (V staging)
  const int vr = tid >> 5;   // 0..7

  uint2 rk[4];
  u32   rv[8];
  auto preload = [&](int kt) {
#pragma unroll
    for (int p = 0; p < 4; ++p)
      rk[p] = *(const uint2*)(qkv + ((size_t)(b * SEQ + kt * 64 + rr + p * 16)) * NQKV
                              + CH + h * DH + cp * 4);
#pragma unroll
    for (int p = 0; p < 8; ++p)
      rv[p] = *(const u32*)(qkv + ((size_t)(b * SEQ + kt * 64 + vr + p * 8)) * NQKV
                            + 2 * CH + h * DH + np * 2);
  };
  preload(0);

  const int KT = SEQ / 64;
  for (int kt = 0; kt < KT; ++kt) {
    __syncthreads();
    // commit staged K chunk
#pragma unroll
    for (int p = 0; p < 4; ++p)
      *(uint2*)(&Ks[(rr + p * 16) * 72 + cp * 4]) = rk[p];
    // commit staged V chunk transposed -> Vt[dv][key]
#pragma unroll
    for (int p = 0; p < 8; ++p) {
      const u32 v = rv[p];
      const int r = vr + p * 8;
      Vt[(np * 2 + 0) * 72 + r] = (u16)(v & 0xFFFFu);
      Vt[(np * 2 + 1) * 72 + r] = (u16)(v >> 16);
    }
    // kick off next chunk's loads (clamped tail), overlap with compute below
    preload((kt + 1 < KT) ? (kt + 1) : kt);
    __syncthreads();

    // S = Q @ K^T for 64 keys: four 16x16 tiles
    FAcc s[4];
#pragma unroll
    for (int hh = 0; hh < 4; ++hh) {
#pragma unroll
      for (int j = 0; j < 8; ++j) s[hh].f[j] = 0.f;
#pragma unroll
      for (int i = 0; i < 2; ++i) {
        AFrag bk;
        const u16* p = &Ks[(hh * 16 + lm) * 72 + i * 32 + kh * 8];
        bk.q[0] = *(const uint4*)(p);
        bk.q[1] = *(const uint4*)(p + 16);
        s[hh].v = wmma_bf16(aq[i].v, bk.v, s[hh].v);
      }
    }

    // online softmax per accumulator slot (row M = j + 8*kh, col N = lm)
#pragma unroll
    for (int j = 0; j < 8; ++j) {
      float sv[4];
#pragma unroll
      for (int hh = 0; hh < 4; ++hh) sv[hh] = s[hh].f[j] * ATT_SCALE;
      float mx = fmaxf(fmaxf(sv[0], sv[1]), fmaxf(sv[2], sv[3]));
      mx = fmaxf(mx, __shfl_xor(mx, 1, 16));
      mx = fmaxf(mx, __shfl_xor(mx, 2, 16));
      mx = fmaxf(mx, __shfl_xor(mx, 4, 16));
      mx = fmaxf(mx, __shfl_xor(mx, 8, 16));
      float mnew = fmaxf(rm[j], mx);
      float pv[4], ps = 0.f;
#pragma unroll
      for (int hh = 0; hh < 4; ++hh) { pv[hh] = __expf(sv[hh] - mnew); ps += pv[hh]; }
      ps += __shfl_xor(ps, 1, 16);
      ps += __shfl_xor(ps, 2, 16);
      ps += __shfl_xor(ps, 4, 16);
      ps += __shfl_xor(ps, 8, 16);
      float alpha = __expf(rm[j] - mnew);
      rl[j] = rl[j] * alpha + ps;
      rm[j] = mnew;
#pragma unroll
      for (int tI = 0; tI < 4; ++tI) oacc[tI].f[j] *= alpha;
      const int prow = j + 8 * kh;
#pragma unroll
      for (int hh = 0; hh < 4; ++hh)
        Pw[prow * 64 + hh * 16 + lm] = f2bf(pv[hh]);
    }
    __asm__ volatile("s_wait_dscnt 0" ::: "memory");

    // reload P in A-fragment layout (16 x 64 -> two k-steps)
    AFrag ap[2];
#pragma unroll
    for (int i = 0; i < 2; ++i) {
      const u16* p = &Pw[lm * 64 + i * 32 + kh * 8];
      ap[i].q[0] = *(const uint4*)(p);
      ap[i].q[1] = *(const uint4*)(p + 16);
    }
    // O += P @ V
#pragma unroll
    for (int tI = 0; tI < 4; ++tI) {
#pragma unroll
      for (int i = 0; i < 2; ++i) {
        AFrag bv;
        const u16* p = &Vt[(tI * 16 + lm) * 72 + i * 32 + kh * 8];
        bv.q[0] = *(const uint4*)(p);
        bv.q[1] = *(const uint4*)(p + 16);
        oacc[tI].v = wmma_bf16(ap[i].v, bv.v, oacc[tI].v);
      }
    }
  }

  // normalize and write bf16 [B*N][768]
#pragma unroll
  for (int j = 0; j < 8; ++j) {
    const float inv = 1.0f / rl[j];
    const size_t row = (size_t)(b * SEQ + q0 + j + 8 * kh);
#pragma unroll
    for (int tI = 0; tI < 4; ++tI) {
      const int col = h * DH + tI * 16 + lm;
      ob[row * CH + col] = f2bf(oacc[tI].f[j] * inv);
    }
  }
}

// ---------------------------------------------------------------- out += 8 * t2 @ B_proj
__global__ void lora_add_kernel(float* __restrict__ out, const float* __restrict__ t2,
                                const float* __restrict__ Bp, size_t total) {
  size_t i = (size_t)blockIdx.x * blockDim.x + threadIdx.x;
  size_t stride = (size_t)gridDim.x * blockDim.x;
  for (; i < total; i += stride) {
    const size_t row = i / CH;
    const int    col = (int)(i % CH);
    const float4 tv = *(const float4*)(t2 + row * 4);
    float v = tv.x * Bp[col] + tv.y * Bp[CH + col] + tv.z * Bp[2 * CH + col] + tv.w * Bp[3 * CH + col];
    out[i] += LORA_SCALE * v;
  }
}

// ---------------------------------------------------------------- launch
extern "C" void kernel_launch(void* const* d_in, const int* in_sizes, int n_in,
                              void* d_out, int out_size, void* d_ws, size_t ws_size,
                              hipStream_t stream) {
  const float* x     = (const float*)d_in[0];
  const float* Wqkv  = (const float*)d_in[1];
  const float* Wproj = (const float*)d_in[2];
  const float* bproj = (const float*)d_in[3];
  const float* Aqkv  = (const float*)d_in[4];
  const float* Bqkv  = (const float*)d_in[5];
  const float* Aproj = (const float*)d_in[6];
  const float* Bproj = (const float*)d_in[7];
  float* out = (float*)d_out;

  char* ws = (char*)d_ws;
  size_t off = 0;
  auto carve = [&](size_t bytes) -> char* {
    char* p = ws + off;
    off = (off + bytes + 255) & ~(size_t)255;
    return p;
  };
  u16*   xb     = (u16*)  carve((size_t)MROWS * CH * 2);
  u16*   wqkvT  = (u16*)  carve((size_t)NQKV * CH * 2);   // [2304][768]
  u16*   wprojT = (u16*)  carve((size_t)CH * CH * 2);     // [768][768]
  u16*   qkvb   = (u16*)  carve((size_t)MROWS * NQKV * 2);
  u16*   obuf   = (u16*)  carve((size_t)MROWS * CH * 2);
  float* t1     = (float*)carve((size_t)MROWS * 4 * 4);
  float* t2     = (float*)carve((size_t)MROWS * 4 * 4);

  // 1. down-convert activations; transpose+convert weights
  {
    size_t n = (size_t)MROWS * CH;
    cast_bf16_kernel<<<(int)((n + 255) / 256), 256, 0, stream>>>(x, xb, n);
    size_t nw = (size_t)CH * NQKV;
    cast_transpose_kernel<<<(int)((nw + 255) / 256), 256, 0, stream>>>(Wqkv, wqkvT, CH, NQKV);
    size_t np = (size_t)CH * CH;
    cast_transpose_kernel<<<(int)((np + 255) / 256), 256, 0, stream>>>(Wproj, wprojT, CH, CH);
  }
  // 2. t1 = x @ A_qkv
  rank4_kernel<<<MROWS, 256, 0, stream>>>(x, Aqkv, t1, CH);
  // 3. qkv = x@W_qkv + 8 * t1@B_qkv   (bf16 out)
  gemm_bf16_kernel<<<dim3(NQKV / 128, MROWS / 128), 256, 0, stream>>>(
      xb, wqkvT, MROWS, NQKV, CH, /*mode=*/0, t1, Bqkv, nullptr, qkvb, nullptr);
  // 4. flash attention
  attn_kernel<<<BATCH * HEADS * (SEQ / 128), 256, 0, stream>>>(qkvb, obuf);
  // 5. out = O @ W_proj + b_proj   (f32 out)
  gemm_bf16_kernel<<<dim3(CH / 128, MROWS / 128), 256, 0, stream>>>(
      obuf, wprojT, MROWS, CH, CH, /*mode=*/1, nullptr, nullptr, bproj, nullptr, out);
  // 6. t2 = out @ A_proj
  rank4_kernel<<<MROWS, 256, 0, stream>>>(out, Aproj, t2, CH);
  // 7. out += 8 * t2 @ B_proj
  {
    size_t total = (size_t)MROWS * CH;
    lora_add_kernel<<<(int)((total + 255) / 256), 256, 0, stream>>>(out, t2, Bproj, total);
  }
}
